// ExpertChoiceRouter_13486197310137
// MI455X (gfx1250) — matrix operand order repaired and checked by reading, not compiled, not run on hardware
//
#include <hip/hip_runtime.h>
#include <hip/hip_bf16.h>

// ---------- problem constants (match reference) ----------
#define NTOK    16384
#define HDIM    2048
#define NEXP    64
#define CAP     320        // int(1.25 * 16384 / 64)
#define SORTN   512        // padded bitonic size >= CAP

// ---------- CDNA5 WMMA types ----------
typedef __attribute__((ext_vector_type(16))) __bf16 v16bf;
typedef __attribute__((ext_vector_type(8)))  float  v8f;

union BF16x16 {
    v16bf v;
    unsigned int u[8];
};

// ---------- helpers ----------
__device__ __forceinline__ unsigned short f2bf(float f) {
    unsigned u = __float_as_uint(f);
    u += 0x7FFFu + ((u >> 16) & 1u);        // round-to-nearest-even
    return (unsigned short)(u >> 16);
}
__device__ __forceinline__ unsigned pk2bf(float lo, float hi) {
    return (unsigned)f2bf(lo) | ((unsigned)f2bf(hi) << 16);
}
// IEEE-754 float -> order-preserving unsigned key
__device__ __forceinline__ unsigned f2key(float f) {
    unsigned u = __float_as_uint(f);
    unsigned m = (unsigned)((int)u >> 31) | 0x80000000u;
    return u ^ m;
}

// ---------- kernel 1: normalize expert embeddings, store bf16 ----------
__global__ void __launch_bounds__(256)
k_expert_norm(const float* __restrict__ emb, unsigned* __restrict__ eb) {
    const int e   = blockIdx.x;           // 64 blocks
    const int tid = threadIdx.x;          // 256 threads, 8 floats each
    const float* row = emb + (size_t)e * HDIM;

    const float4* p = (const float4*)(row + tid * 8);
    float4 a = p[0], b = p[1];
    float ss = a.x*a.x + a.y*a.y + a.z*a.z + a.w*a.w
             + b.x*b.x + b.y*b.y + b.z*b.z + b.w*b.w;

    __shared__ float red[256];
    red[tid] = ss;
    __syncthreads();
    for (int s = 128; s > 0; s >>= 1) {
        if (tid < s) red[tid] += red[tid + s];
        __syncthreads();
    }
    const float scale = 1.0f / fmaxf(sqrtf(red[0]), 1e-12f);

    uint4 o;
    o.x = pk2bf(a.x * scale, a.y * scale);
    o.y = pk2bf(a.z * scale, a.w * scale);
    o.z = pk2bf(b.x * scale, b.y * scale);
    o.w = pk2bf(b.z * scale, b.w * scale);
    ((uint4*)(eb + (size_t)e * (HDIM / 2)))[tid] = o;
}

// ---------- kernel 2: zero combine_weights ----------
__global__ void k_zero(float* __restrict__ p, int n) {
    int i = blockIdx.x * blockDim.x + threadIdx.x;
    if (i < n) p[i] = 0.0f;
}

// ---------- kernel 3: fused affinity GEMM (bf16 WMMA) + column 1/||h|| ----------
// grid: NTOK/64 blocks of 128 threads (4 waves); each wave -> 16 tokens x 64 experts
__global__ void __launch_bounds__(128)
k_affinity(const float* __restrict__ hidden, const unsigned* __restrict__ eb,
           float* __restrict__ aff) {
    const int lane = threadIdx.x & 31;
    const int wave = threadIdx.x >> 5;
    const int ncol = lane & 15;           // B/C column within tile
    const int hi   = lane >> 4;           // which K-half this lane covers
    const int token = blockIdx.x * 64 + wave * 16 + ncol;

    const float* hrow = hidden + (size_t)token * HDIM;

    v8f acc0 = {}, acc1 = {}, acc2 = {}, acc3 = {};
    float ss = 0.0f;
    const unsigned mrow = (unsigned)(lane & 15);   // expert row within M-tile

    for (int kc = 0; kc < HDIM; kc += 32) {
        // prefetch next chunk of this token row (-> global_prefetch_b8)
        __builtin_prefetch(hrow + kc + 256, 0, 1);

        // ---- B fragment: 16 contiguous K (f32) for this lane's K-half ----
        const float4* bp = (const float4*)(hrow + kc + hi * 16);
        float4 x0 = bp[0], x1 = bp[1], x2 = bp[2], x3 = bp[3];
        ss += x0.x*x0.x + x0.y*x0.y + x0.z*x0.z + x0.w*x0.w
            + x1.x*x1.x + x1.y*x1.y + x1.z*x1.z + x1.w*x1.w
            + x2.x*x2.x + x2.y*x2.y + x2.z*x2.z + x2.w*x2.w
            + x3.x*x3.x + x3.y*x3.y + x3.z*x3.z + x3.w*x3.w;

        BF16x16 B;
        B.u[0] = pk2bf(x0.x, x0.y);  B.u[1] = pk2bf(x0.z, x0.w);
        B.u[2] = pk2bf(x1.x, x1.y);  B.u[3] = pk2bf(x1.z, x1.w);
        B.u[4] = pk2bf(x2.x, x2.y);  B.u[5] = pk2bf(x2.z, x2.w);
        B.u[6] = pk2bf(x3.x, x3.y);  B.u[7] = pk2bf(x3.z, x3.w);

        // ---- A fragments: 4 expert tiles (M = 0..63) ----
        #pragma unroll
        for (int t = 0; t < 4; ++t) {
            const unsigned m = (unsigned)t * 16u + mrow;
            // lane needs K-locals [hi*16 .. hi*16+15]
            const uint4* ap =
                (const uint4*)(eb + (size_t)m * (HDIM / 2) + (kc >> 1) + hi * 8);
            uint4 a0 = ap[0];        // first  8 bf16
            uint4 a1 = ap[1];        // +4 uints = +8 bf16
            BF16x16 A;
            A.u[0] = a0.x; A.u[1] = a0.y; A.u[2] = a0.z; A.u[3] = a0.w;
            A.u[4] = a1.x; A.u[5] = a1.y; A.u[6] = a1.z; A.u[7] = a1.w;

            if (t == 0)
                acc0 = __builtin_amdgcn_wmma_f32_16x16x32_bf16(
                        false, A.v, false, B.v, (short)0, acc0, false, false);
            else if (t == 1)
                acc1 = __builtin_amdgcn_wmma_f32_16x16x32_bf16(
                        false, A.v, false, B.v, (short)0, acc1, false, false);
            else if (t == 2)
                acc2 = __builtin_amdgcn_wmma_f32_16x16x32_bf16(
                        false, A.v, false, B.v, (short)0, acc2, false, false);
            else
                acc3 = __builtin_amdgcn_wmma_f32_16x16x32_bf16(
                        false, A.v, false, B.v, (short)0, acc3, false, false);
        }
    }

    // lanes (l, l+16) together cover the full K range of one token
    float ss_all = ss + __shfl_xor(ss, 16, 32);
    const float scale = 1.0f / fmaxf(sqrtf(ss_all), 1e-12f);

    // C layout: VGPR v -> expert row (v + 8*hi) within tile, column = ncol
    #pragma unroll
    for (int v = 0; v < 8; ++v) {
        const int r = v + 8 * hi;
        aff[(size_t)(0 * 16 + r) * NTOK + token] = acc0[v] * scale;
        aff[(size_t)(1 * 16 + r) * NTOK + token] = acc1[v] * scale;
        aff[(size_t)(2 * 16 + r) * NTOK + token] = acc2[v] * scale;
        aff[(size_t)(3 * 16 + r) * NTOK + token] = acc3[v] * scale;
    }
}

// ---------- kernel 4: per-expert exact top-320 (radix select) + sort + softmax ----------
__global__ void __launch_bounds__(256)
k_topk(const float* __restrict__ aff, float* __restrict__ outW,
       int* __restrict__ outI, float* __restrict__ combine) {
    const int e   = blockIdx.x;          // 64 blocks
    const int tid = threadIdx.x;         // 256 threads
    const float* row = aff + (size_t)e * NTOK;

    __shared__ unsigned hist[256];
    __shared__ float    selV[SORTN];
    __shared__ int      selI[SORTN];
    __shared__ unsigned sh_prefix, sh_remaining, sh_cntG, sh_cntE;
    __shared__ float    red[256];

    // ---- 4-pass MSB-first radix select: exact key of 320th largest ----
    unsigned prefix = 0, mask = 0, remaining = CAP;
    for (int pos = 3; pos >= 0; --pos) {
        hist[tid] = 0;
        __syncthreads();
        for (int i = tid; i < NTOK; i += 256) {
            unsigned k = f2key(row[i]);
            if ((k & mask) == prefix)
                atomicAdd(&hist[(k >> (pos * 8)) & 255u], 1u);
        }
        __syncthreads();
        if (tid == 0) {
            unsigned cum = 0; int b = 0;
            for (int j = 255; j >= 0; --j) {
                unsigned c = hist[j];
                if (cum + c >= remaining) { b = j; break; }
                cum += c;
            }
            sh_prefix    = prefix | ((unsigned)b << (pos * 8));
            sh_remaining = remaining - cum;
        }
        __syncthreads();
        prefix    = sh_prefix;
        remaining = sh_remaining;
        mask     |= 0xFFu << (pos * 8);
        __syncthreads();
    }
    const unsigned T      = prefix;          // key of the 320th largest
    const unsigned needEq = remaining;       // how many ==T to take
    const unsigned cntGfx = CAP - needEq;    // count strictly greater

    // ---- collect ----
    if (tid == 0) { sh_cntG = 0; sh_cntE = 0; }
    __syncthreads();
    for (int i = tid; i < NTOK; i += 256) {
        float    v = row[i];
        unsigned k = f2key(v);
        if (k > T) {
            unsigned p = atomicAdd(&sh_cntG, 1u);
            selV[p] = v; selI[p] = i;
        } else if (k == T) {
            unsigned q = atomicAdd(&sh_cntE, 1u);
            if (q < needEq) { unsigned p = cntGfx + q; selV[p] = v; selI[p] = i; }
        }
    }
    for (int i = CAP + tid; i < SORTN; i += 256) {
        selV[i] = -__builtin_inff(); selI[i] = 0;
    }
    __syncthreads();

    // ---- bitonic sort descending (512 elems, 256 threads) ----
    for (unsigned sz = 2; sz <= SORTN; sz <<= 1) {
        for (unsigned j = sz >> 1; j > 0; j >>= 1) {
            for (unsigned i = tid; i < SORTN; i += 256) {
                unsigned ixj = i ^ j;
                if (ixj > i) {
                    bool desc = ((i & sz) == 0);
                    bool sw = desc ? (selV[i] < selV[ixj]) : (selV[i] > selV[ixj]);
                    if (sw) {
                        float tv = selV[i]; selV[i] = selV[ixj]; selV[ixj] = tv;
                        int   ti = selI[i]; selI[i] = selI[ixj]; selI[ixj] = ti;
                    }
                }
            }
            __syncthreads();
        }
    }

    // ---- softmax over the 320 sorted weights ----
    const float mx = selV[0];
    __syncthreads();
    float local = 0.0f;
    for (int c = tid; c < CAP; c += 256) {
        float ex = __expf(selV[c] - mx);
        selV[c] = ex;
        local  += ex;
    }
    red[tid] = local;
    __syncthreads();
    for (int s = 128; s > 0; s >>= 1) {
        if (tid < s) red[tid] += red[tid + s];
        __syncthreads();
    }
    const float inv = 1.0f / red[0];

    // ---- outputs: weights, indices, scatter into combine ----
    for (int c = tid; c < CAP; c += 256) {
        float w   = selV[c] * inv;
        int   tok = selI[c];
        outW[(size_t)e * CAP + c] = w;
        outI[(size_t)e * CAP + c] = tok;
        combine[(size_t)tok * NEXP + e] = w;   // unique (token,expert) writer
    }
}

// ---------- launch ----------
extern "C" void kernel_launch(void* const* d_in, const int* in_sizes, int n_in,
                              void* d_out, int out_size, void* d_ws, size_t ws_size,
                              hipStream_t stream) {
    (void)in_sizes; (void)n_in; (void)out_size; (void)ws_size;

    const float* hidden = (const float*)d_in[0];   // [16384, 2048] f32
    const float* emb    = (const float*)d_in[1];   // [64, 2048]   f32

    // workspace: [bf16 normalized experts 256KB][affinity f32 4MB]
    unsigned char* ws  = (unsigned char*)d_ws;
    unsigned*      eb  = (unsigned*)ws;
    float*         aff = (float*)(ws + (size_t)NEXP * HDIM * 2);

    // outputs (flat 4-byte concat): weights[64*320] | indices[64*320] | combine[16384*64]
    float* outW    = (float*)d_out;
    int*   outI    = (int*)((float*)d_out + (size_t)NEXP * CAP);
    float* combine = (float*)d_out + (size_t)2 * NEXP * CAP;

    k_expert_norm<<<NEXP, 256, 0, stream>>>(emb, eb);
    {
        const int n = NTOK * NEXP;
        k_zero<<<(n + 255) / 256, 256, 0, stream>>>(combine, n);
    }
    k_affinity<<<NTOK / 64, 128, 0, stream>>>(hidden, eb, aff);
    k_topk<<<NEXP, 256, 0, stream>>>(aff, outW, outI, combine);
}
